// QuantumAttention_74603581932037
// MI455X (gfx1250) — matrix operand order repaired and checked by reading, not compiled, not run on hardware
//
#include <hip/hip_runtime.h>
#include <math.h>

typedef unsigned short u16;
typedef unsigned int   u32;
typedef __attribute__((ext_vector_type(16))) __bf16 v16bf;
typedef __attribute__((ext_vector_type(8)))  __bf16 v8bf;
typedef __attribute__((ext_vector_type(8)))  float  v8f;

#define BB   2
#define SEQ  2048
#define NH   16
#define HD   128
#define HIDN 2048
#define MR   (BB*SEQ)   // 4096 rows (B*S)

// ---------- helpers ----------
__device__ __forceinline__ u32 pk2bf(float a, float b){
  u32 ua = __float_as_uint(a), ub = __float_as_uint(b);
  ua = (ua + 0x7FFFu + ((ua >> 16) & 1u)) >> 16;
  ub = (ub + 0x7FFFu + ((ub >> 16) & 1u)) >> 16;
  return ua | (ub << 16);
}
__device__ __forceinline__ u16 f2bf(float a){
  u32 ua = __float_as_uint(a);
  return (u16)((ua + 0x7FFFu + ((ua >> 16) & 1u)) >> 16);
}
__device__ __forceinline__ v8f vzero8(){
  v8f v;
  #pragma unroll
  for (int i = 0; i < 8; ++i) v[i] = 0.0f;
  return v;
}
__device__ __forceinline__ v16bf combine16(v8bf a, v8bf b){
  v16bf r;
  #pragma unroll
  for (int i = 0; i < 8; ++i) { r[i] = a[i]; r[i+8] = b[i]; }
  return r;
}
__device__ __forceinline__ v16bf load_pair(const u16* p0, const u16* p1){
  return combine16(*(const v8bf*)p0, *(const v8bf*)p1);
}
// A fragment 16(M)x32(K) from row-major [*, ld] bf16.
// Contiguous k order (same as B): lanes 0-15 row M=lane, K=0..15; lanes 16-31 row M=lane-16, K=16..31
__device__ __forceinline__ v16bf load_afrag(const u16* base, int ld, int row0, int k0){
  int lane = (int)(threadIdx.x & 31u);
  int hf = lane >> 4, r = lane & 15;
  const u16* p = base + (size_t)(row0 + r) * ld + (k0 + hf * 16);
  return load_pair(p, p + 8);
}
// B fragment 32(K)x16(N); column n sourced from row n of a row-major [N, K] matrix.
// Contiguous k order (same as A): lanes 0-15 col N=lane, K=0..15; lanes 16-31 col N=lane-16, K=16..31
__device__ __forceinline__ v16bf load_bfrag(const u16* base, int ld, int n0, int k0){
  int lane = (int)(threadIdx.x & 31u);
  int hf = lane >> 4, n = lane & 15;
  const u16* p = base + (size_t)(n0 + n) * ld + (k0 + hf * 16);
  return load_pair(p, p + 8);
}
__device__ __forceinline__ v8f wmma_bf16(v16bf a, v16bf b, v8f c){
  return __builtin_amdgcn_wmma_f32_16x16x32_bf16(false, a, false, b, (short)0, c, false, false);
}

// wave-level 64x64 GEMM: C[m0:m0+64, n0:n0+64] += A[M,K] * W[N,K]^T (both row-major bf16)
__device__ __forceinline__ void gemm_wave64(const u16* __restrict__ A, const u16* __restrict__ W,
                                            int K, int m0, int n0, v8f (&acc)[4][4]){
  for (int k0 = 0; k0 < K; k0 += 32){
    v16bf af[4], bf[4];
    #pragma unroll
    for (int i = 0; i < 4; ++i) { af[i] = load_afrag(A, K, m0 + i*16, k0); }
    #pragma unroll
    for (int j = 0; j < 4; ++j) { bf[j] = load_bfrag(W, K, n0 + j*16, k0); }
    #pragma unroll
    for (int i = 0; i < 4; ++i){
      #pragma unroll
      for (int j = 0; j < 4; ++j){
        acc[i][j] = wmma_bf16(af[i], bf[j], acc[i][j]);
      }
    }
  }
}

// ---------- conversion / prep ----------
__global__ void __launch_bounds__(256) cvt_bf16_kernel(const float* __restrict__ src,
                                                       u16* __restrict__ dst, int n){
  int i = (int)(blockIdx.x * 256 + threadIdx.x) * 2;
  if (i < n) ((u32*)dst)[i >> 1] = pk2bf(src[i], src[i + 1]);
}

__global__ void __launch_bounds__(256) qscale_kernel(
    const float* __restrict__ qr, const float* __restrict__ qi, const float* __restrict__ qm,
    const float* __restrict__ kr, const float* __restrict__ ki, const float* __restrict__ km,
    const float* __restrict__ vr, const float* __restrict__ vi, const float* __restrict__ vm,
    float* __restrict__ sq, float* __restrict__ sk, float* __restrict__ sv){
  int i = blockIdx.x * 256 + threadIdx.x;
  if (i < HIDN){
    sq[i] = sqrtf(qr[i]*qr[i] + qi[i]*qi[i]) * fabsf(qm[i]);
    sk[i] = sqrtf(kr[i]*kr[i] + ki[i]*ki[i]) * fabsf(km[i]);
    sv[i] = sqrtf(vr[i]*vr[i] + vi[i]*vi[i]) * fabsf(vm[i]);
  }
}

// ---------- phase 1: QKV projection + quantum epilogue ----------
__global__ void __launch_bounds__(256) qkv_gemm_kernel(
    const u16* __restrict__ X,
    const u16* __restrict__ Wq, const u16* __restrict__ Wk, const u16* __restrict__ Wv,
    const float* __restrict__ bq, const float* __restrict__ bk, const float* __restrict__ bv,
    const float* __restrict__ sq, const float* __restrict__ sk, const float* __restrict__ sv,
    u16* __restrict__ qo, u16* __restrict__ ko, u16* __restrict__ vto)
{
  int z = blockIdx.z;
  const u16* W      = (z == 0) ? Wq : (z == 1) ? Wk : Wv;
  const float* bias = (z == 0) ? bq : (z == 1) ? bk : bv;
  const float* scl  = (z == 0) ? sq : (z == 1) ? sk : sv;
  int wave = (int)(threadIdx.x >> 5);
  int wr = wave >> 2, wc = wave & 3;
  int m0 = blockIdx.x * 128 + wr * 64;
  int n0 = blockIdx.y * 256 + wc * 64;
  v8f acc[4][4];
  #pragma unroll
  for (int i = 0; i < 4; ++i){
    #pragma unroll
    for (int j = 0; j < 4; ++j) acc[i][j] = vzero8();
  }
  gemm_wave64(X, W, HIDN, m0, n0, acc);

  int lane = threadIdx.x & 31, hf = lane >> 4, nl = lane & 15;
  #pragma unroll
  for (int i = 0; i < 4; ++i){
    #pragma unroll
    for (int j = 0; j < 4; ++j){
      int mb = m0 + i*16 + hf*8;         // first of 8 consecutive rows
      int nn = n0 + j*16 + nl;           // output channel (fixed per lane)
      float sc = scl[nn], bi = bias[nn];
      int bidx = mb >> 11, s0 = mb & (SEQ - 1);
      int hh = nn >> 7,   dd = nn & (HD - 1);
      float vals[8];
      #pragma unroll
      for (int r = 0; r < 8; ++r) vals[r] = fabsf(acc[i][j][r] + bi) * sc;
      if (z < 2){
        u16* dst = (z == 0) ? qo : ko;   // [B,H,S,D]
        size_t basep = ((size_t)(bidx*NH + hh) * SEQ) * HD + dd;
        #pragma unroll
        for (int r = 0; r < 8; ++r) dst[basep + (size_t)(s0 + r) * HD] = f2bf(vals[r]);
      } else {                           // V stored transposed: [B,H,D,S], s contiguous
        size_t off = ((size_t)(bidx*NH + hh) * HD + dd) * SEQ + s0;
        uint4 pk;
        pk.x = pk2bf(vals[0], vals[1]); pk.y = pk2bf(vals[2], vals[3]);
        pk.z = pk2bf(vals[4], vals[5]); pk.w = pk2bf(vals[6], vals[7]);
        *(uint4*)(vto + off) = pk;
      }
    }
  }
}

// ---------- phase 2: flash attention (transposed S and ctx) ----------
__global__ void __launch_bounds__(128) flash_attn_kernel(
    const u16* __restrict__ qb, const u16* __restrict__ kb, const u16* __restrict__ vtb,
    const float* __restrict__ mask, const float* __restrict__ ent,
    u16* __restrict__ ctxb)
{
  __shared__ u16 P[4][32][32];           // per-wave P tile: [q-row][kv], bf16
  int tid  = threadIdx.x;
  int wave = tid >> 5;
  int lane = tid & 31;
  int hf = lane >> 4;
  int nl = lane & 15;
  int bh = blockIdx.y;
  int b  = bh >> 4;
  int h  = bh & 15;
  int q0 = blockIdx.x * 128 + wave * 32;

  const u16* Q  = qb  + (size_t)bh * SEQ * HD;   // [S,128]
  const u16* Kp = kb  + (size_t)bh * SEQ * HD;   // [S,128]
  const u16* Vt = vtb + (size_t)bh * HD * SEQ;   // [128,S]
  const float* mrow = mask + (size_t)b * SEQ;

  // resident Q^T B-fragments: 2 q-subtiles x 4 d-chunks
  v16bf qf[2][4];
  #pragma unroll
  for (int j = 0; j < 2; ++j){
    #pragma unroll
    for (int ks = 0; ks < 4; ++ks) qf[j][ks] = load_bfrag(Q, HD, q0 + j*16, ks*32);
  }

  // ent bias: kv%16 == hf*8+r for 16-aligned kv tiles -> 8 lane constants
  float be[8];
  #pragma unroll
  for (int r = 0; r < 8; ++r) be[r] = ent[h*NH + hf*8 + r];

  float mrun[2] = {-INFINITY, -INFINITY};
  float lrun[2] = {0.0f, 0.0f};
  v8f ctxa[8][2];                        // ctx^T accumulators: 8 d-tiles x 2 q-tiles
  #pragma unroll
  for (int i = 0; i < 8; ++i){
    #pragma unroll
    for (int j = 0; j < 2; ++j) ctxa[i][j] = vzero8();
  }

  const float rsD = 0.08838834764831845f;  // 1/sqrt(128)

  for (int kv0 = 0; kv0 < SEQ; kv0 += 32){
    // S^T chunk = K[kv0:kv0+32, :] * Q^T  -> [32 kv x 32 q]
    v8f st[2][2];
    #pragma unroll
    for (int t = 0; t < 2; ++t){
      #pragma unroll
      for (int j = 0; j < 2; ++j) st[t][j] = vzero8();
    }
    #pragma unroll
    for (int ks = 0; ks < 4; ++ks){
      v16bf kf0 = load_afrag(Kp, HD, kv0,      ks*32);
      v16bf kf1 = load_afrag(Kp, HD, kv0 + 16, ks*32);
      #pragma unroll
      for (int j = 0; j < 2; ++j){
        st[0][j] = wmma_bf16(kf0, qf[j][ks], st[0][j]);
        st[1][j] = wmma_bf16(kf1, qf[j][ks], st[1][j]);
      }
    }
    // scale + bias + mask
    float sv[2][2][8];
    #pragma unroll
    for (int t = 0; t < 2; ++t){
      const float4* mp = (const float4*)(mrow + kv0 + t*16 + hf*8);
      float4 ma = mp[0], mb4 = mp[1];
      float mk[8] = {ma.x, ma.y, ma.z, ma.w, mb4.x, mb4.y, mb4.z, mb4.w};
      #pragma unroll
      for (int j = 0; j < 2; ++j){
        #pragma unroll
        for (int r = 0; r < 8; ++r) sv[t][j][r] = st[t][j][r] * rsD + be[r] + mk[r];
      }
    }
    // online softmax per q column (lane n & n+16 hold complementary kv halves)
    #pragma unroll
    for (int j = 0; j < 2; ++j){
      float cmax = -INFINITY;
      #pragma unroll
      for (int t = 0; t < 2; ++t){
        #pragma unroll
        for (int r = 0; r < 8; ++r) cmax = fmaxf(cmax, sv[t][j][r]);
      }
      cmax = fmaxf(cmax, __shfl_xor(cmax, 16));
      float mnew  = fmaxf(mrun[j], cmax);
      float alpha = __expf(mrun[j] - mnew);
      mrun[j] = mnew;
      float ls = 0.0f;
      #pragma unroll
      for (int t = 0; t < 2; ++t){
        #pragma unroll
        for (int r = 0; r < 8; ++r){
          float p = __expf(sv[t][j][r] - mnew);
          sv[t][j][r] = p;
          ls += p;
        }
      }
      ls += __shfl_xor(ls, 16);
      lrun[j] = lrun[j] * alpha + ls;
      #pragma unroll
      for (int i = 0; i < 8; ++i){
        #pragma unroll
        for (int r = 0; r < 8; ++r) ctxa[i][j][r] *= alpha;
      }
      // pack P into LDS as [q][kv] (8 contiguous kv per lane)
      #pragma unroll
      for (int t = 0; t < 2; ++t){
        uint4 pk;
        pk.x = pk2bf(sv[t][j][0], sv[t][j][1]);
        pk.y = pk2bf(sv[t][j][2], sv[t][j][3]);
        pk.z = pk2bf(sv[t][j][4], sv[t][j][5]);
        pk.w = pk2bf(sv[t][j][6], sv[t][j][7]);
        *(uint4*)&P[wave][j*16 + nl][t*16 + hf*8] = pk;
      }
    }
    __syncthreads();
    // ctx^T += V^T[d, kv0:kv0+32] * P^T
    #pragma unroll
    for (int i = 0; i < 8; ++i){
      v16bf vf = load_afrag(Vt, SEQ, i*16, kv0);
      #pragma unroll
      for (int j = 0; j < 2; ++j){
        const u16* pp = &P[wave][j*16 + nl][hf*16];
        v16bf pf = load_pair(pp, pp + 8);
        ctxa[i][j] = wmma_bf16(vf, pf, ctxa[i][j]);
      }
    }
    __syncthreads();
  }

  // normalize & store ctx as [B,S,HID] bf16 (d contiguous per lane)
  #pragma unroll
  for (int j = 0; j < 2; ++j){
    float inv = 1.0f / lrun[j];
    int s = q0 + j*16 + nl;
    size_t rowoff = ((size_t)(b * SEQ + s)) * HIDN + h * HD;
    #pragma unroll
    for (int i = 0; i < 8; ++i){
      float vals[8];
      #pragma unroll
      for (int r = 0; r < 8; ++r) vals[r] = ctxa[i][j][r] * inv;
      uint4 pk;
      pk.x = pk2bf(vals[0], vals[1]); pk.y = pk2bf(vals[2], vals[3]);
      pk.z = pk2bf(vals[4], vals[5]); pk.w = pk2bf(vals[6], vals[7]);
      *(uint4*)(ctxb + rowoff + i*16 + hf*8) = pk;
    }
  }
}

// ---------- phase 3: output projection ----------
__global__ void __launch_bounds__(256) oproj_kernel(
    const u16* __restrict__ Ctx, const u16* __restrict__ Wo,
    const float* __restrict__ bo, float* __restrict__ out)
{
  int wave = (int)(threadIdx.x >> 5);
  int wr = wave >> 2, wc = wave & 3;
  int m0 = blockIdx.x * 128 + wr * 64;
  int n0 = blockIdx.y * 256 + wc * 64;
  v8f acc[4][4];
  #pragma unroll
  for (int i = 0; i < 4; ++i){
    #pragma unroll
    for (int j = 0; j < 4; ++j) acc[i][j] = vzero8();
  }
  gemm_wave64(Ctx, Wo, HIDN, m0, n0, acc);

  int lane = threadIdx.x & 31, hf = lane >> 4, nl = lane & 15;
  #pragma unroll
  for (int i = 0; i < 4; ++i){
    #pragma unroll
    for (int j = 0; j < 4; ++j){
      int mb = m0 + i*16 + hf*8;
      int nn = n0 + j*16 + nl;
      float bi = bo[nn];
      #pragma unroll
      for (int r = 0; r < 8; ++r)
        out[(size_t)(mb + r) * HIDN + nn] = acc[i][j][r] + bi;
    }
  }
}

// ---------- launcher ----------
extern "C" void kernel_launch(void* const* d_in, const int* in_sizes, int n_in,
                              void* d_out, int out_size, void* d_ws, size_t ws_size,
                              hipStream_t stream)
{
  const float* hidden = (const float*)d_in[0];
  const float* mask   = (const float*)d_in[1];
  const float* q_real = (const float*)d_in[2];
  const float* q_imag = (const float*)d_in[3];
  const float* q_mag  = (const float*)d_in[5];
  const float* k_real = (const float*)d_in[6];
  const float* k_imag = (const float*)d_in[7];
  const float* k_mag  = (const float*)d_in[9];
  const float* v_real = (const float*)d_in[10];
  const float* v_imag = (const float*)d_in[11];
  const float* v_mag  = (const float*)d_in[13];
  const float* Wq = (const float*)d_in[14];
  const float* bq = (const float*)d_in[15];
  const float* Wk = (const float*)d_in[16];
  const float* bk = (const float*)d_in[17];
  const float* Wv = (const float*)d_in[18];
  const float* bv = (const float*)d_in[19];
  const float* Wo = (const float*)d_in[20];
  const float* bo = (const float*)d_in[21];
  const float* ent = (const float*)d_in[22];

  char* ws = (char*)d_ws;
  size_t off = 0;
  auto carve = [&](size_t bytes) -> char* {
    char* p = ws + off;
    off += (bytes + 255) & ~(size_t)255;
    return p;
  };
  u16* Xb     = (u16*)carve((size_t)MR   * HIDN * 2);
  u16* Wqb    = (u16*)carve((size_t)HIDN * HIDN * 2);
  u16* Wkb    = (u16*)carve((size_t)HIDN * HIDN * 2);
  u16* Wvb    = (u16*)carve((size_t)HIDN * HIDN * 2);
  u16* Wob    = (u16*)carve((size_t)HIDN * HIDN * 2);
  u16* qbuf   = (u16*)carve((size_t)BB * NH * SEQ * HD * 2);
  u16* kbuf   = (u16*)carve((size_t)BB * NH * SEQ * HD * 2);
  u16* vtbuf  = (u16*)carve((size_t)BB * NH * HD * SEQ * 2);
  u16* ctxbuf = (u16*)carve((size_t)MR * HIDN * 2);
  float* sq   = (float*)carve(HIDN * 4);
  float* sk   = (float*)carve(HIDN * 4);
  float* sv   = (float*)carve(HIDN * 4);

  int nX = MR * HIDN;
  cvt_bf16_kernel<<<nX/512, 256, 0, stream>>>(hidden, Xb, nX);
  int nW = HIDN * HIDN;
  cvt_bf16_kernel<<<nW/512, 256, 0, stream>>>(Wq, Wqb, nW);
  cvt_bf16_kernel<<<nW/512, 256, 0, stream>>>(Wk, Wkb, nW);
  cvt_bf16_kernel<<<nW/512, 256, 0, stream>>>(Wv, Wvb, nW);
  cvt_bf16_kernel<<<nW/512, 256, 0, stream>>>(Wo, Wob, nW);
  qscale_kernel<<<HIDN/256, 256, 0, stream>>>(q_real, q_imag, q_mag,
                                              k_real, k_imag, k_mag,
                                              v_real, v_imag, v_mag, sq, sk, sv);

  dim3 g1(MR/128, HIDN/256, 3);
  qkv_gemm_kernel<<<g1, 256, 0, stream>>>(Xb, Wqb, Wkb, Wvb, bq, bk, bv,
                                          sq, sk, sv, qbuf, kbuf, vtbuf);

  dim3 g2(SEQ/128, BB*NH);
  flash_attn_kernel<<<g2, 128, 0, stream>>>(qbuf, kbuf, vtbuf, mask, ent, ctxbuf);

  dim3 g3(MR/128, HIDN/256);
  oproj_kernel<<<g3, 256, 0, stream>>>(ctxbuf, Wob, bo, (float*)d_out);

  (void)in_sizes; (void)n_in; (void)out_size; (void)ws_size;
}
